// hmodel_24532853195394
// MI455X (gfx1250) — matrix operand order repaired and checked by reading, not compiled, not run on hardware
//
#include <hip/hip_runtime.h>
#include <hip/hip_bf16.h>

#define NPTS 100000
#define CDIM 1024
#define DDIM 256
#define TILE_C 32                     // centroid rows per TDM transfer
#define NTILES (CDIM / TILE_C)        // 32
#define ROW_US 264                    // padded LDS row pitch in ushorts (528 B)
#define WAVES_PER_BLOCK 4

typedef __attribute__((ext_vector_type(16))) __bf16 v16bf;
typedef __attribute__((ext_vector_type(8)))  float  v8f;
typedef __attribute__((ext_vector_type(4)))  unsigned int u32x4;
typedef __attribute__((ext_vector_type(8)))  unsigned int u32x8;

__device__ __forceinline__ unsigned short f32_to_bf16_rne(float f) {
    union { float f; unsigned u; } c; c.f = f;
    unsigned u = c.u;
    unsigned r = u + 0x7FFFu + ((u >> 16) & 1u);   // round-to-nearest-even
    return (unsigned short)(r >> 16);
}

// ---------------------------------------------------------------------------
// TDM: issue a 2D tile load (TILE_C rows x 256 bf16) from global Phi into LDS,
// with LDS padding of 4 DWORDs every 128 DWORDs (row pitch 528 B) so that the
// consumers' ds_load_b128 pattern is bank-conflict free.
// D# layout per CDNA5 ISA (groups 0 and 1 only,
// VADDR2/VADDR3 = NULL for a 2D tensor).
// ---------------------------------------------------------------------------
__device__ __forceinline__ void tdm_load_tile(const unsigned short* gptr,
                                              unsigned lds_off) {
    const unsigned long long ga = (unsigned long long)(uintptr_t)gptr;
    u32x4 g0;
    g0[0] = 1u;                                          // count=1, user D#
    g0[1] = lds_off;                                     // lds_addr
    g0[2] = (unsigned)(ga & 0xFFFFFFFFu);                // global_addr lo
    g0[3] = (unsigned)((ga >> 32) & 0x01FFFFFFu)         // global_addr hi (57b)
          | (2u << 30);                                  // type=2 ("image")
    u32x8 g1;
    g1[0] = (1u << 16)                                   // data_size = 2 bytes
          | (1u << 20)                                   // pad_enable
          | (6u << 22)                                   // pad_interval: 128 DW
          | (3u << 25);                                  // pad_amount: 4 DW
    g1[1] = ((unsigned)DDIM & 0xFFFFu) << 16;            // tensor_dim0 lo16
    g1[2] = ((unsigned)CDIM << 16);                      // dim0 hi=0 | tensor_dim1 lo16
    g1[3] = ((unsigned)DDIM << 16);                      // dim1 hi=0 | tile_dim0=256
    g1[4] = (unsigned)TILE_C;                            // tile_dim1 | tile_dim2=0
    g1[5] = (unsigned)DDIM;                              // tensor_dim0_stride lo32
    g1[6] = 0u;                                          // stride0 hi | stride1 lo
    g1[7] = 0u;                                          // stride1 hi
    asm volatile("tensor_load_to_lds %0, %1" :: "s"(g0), "s"(g1) : "memory");
}

// ---------------------------------------------------------------------------
// Kernel 1: Phi = matrix_parents @ epsilon  (f32 accumulate for precision),
// fused p2[c] = ||Phi[c,:]||^2, Phi stored as bf16 for the WMMA main GEMM.
// ---------------------------------------------------------------------------
#define PHI_ROWS 4
__global__ void __launch_bounds__(256)
phi_p2_kernel(const float* __restrict__ mp, const float* __restrict__ eps,
              unsigned short* __restrict__ phi_bf, float* __restrict__ p2) {
    const int n  = threadIdx.x;
    const int r0 = blockIdx.x * PHI_ROWS;

    const float* m0 = mp + (size_t)r0 * CDIM;
    const float* m1 = m0 + CDIM;
    const float* m2 = m1 + CDIM;
    const float* m3 = m2 + CDIM;

    float a0 = 0.f, a1 = 0.f, a2 = 0.f, a3 = 0.f;
#pragma unroll 8
    for (int k = 0; k < CDIM; ++k) {
        const float e = eps[(size_t)k * DDIM + n];
        a0 = fmaf(m0[k], e, a0);
        a1 = fmaf(m1[k], e, a1);
        a2 = fmaf(m2[k], e, a2);
        a3 = fmaf(m3[k], e, a3);
    }

    phi_bf[(size_t)(r0 + 0) * DDIM + n] = f32_to_bf16_rne(a0);
    phi_bf[(size_t)(r0 + 1) * DDIM + n] = f32_to_bf16_rne(a1);
    phi_bf[(size_t)(r0 + 2) * DDIM + n] = f32_to_bf16_rne(a2);
    phi_bf[(size_t)(r0 + 3) * DDIM + n] = f32_to_bf16_rne(a3);

    __shared__ float red[256][PHI_ROWS];
    red[n][0] = a0 * a0; red[n][1] = a1 * a1;
    red[n][2] = a2 * a2; red[n][3] = a3 * a3;
    __syncthreads();
    for (int s = 128; s > 0; s >>= 1) {
        if (n < s) {
            red[n][0] += red[n + s][0];
            red[n][1] += red[n + s][1];
            red[n][2] += red[n + s][2];
            red[n][3] += red[n + s][3];
        }
        __syncthreads();
    }
    if (n == 0) {
        p2[r0 + 0] = red[0][0];
        p2[r0 + 1] = red[0][1];
        p2[r0 + 2] = red[0][2];
        p2[r0 + 3] = red[0][3];
    }
}

// ---------------------------------------------------------------------------
// Kernel 2: fused dots = X @ Phi^T (bf16 WMMA) + argmin. Phi tiles staged in
// LDS by the Tensor Data Mover (double buffered, wave 0 issues, TENSORcnt +
// workgroup barriers for hand-off). Two independent WMMA accumulator chains.
// ---------------------------------------------------------------------------
__global__ void __launch_bounds__(32 * WAVES_PER_BLOCK)
cdist_argmin_kernel(const float* __restrict__ X,
                    const unsigned short* __restrict__ phi_bf,
                    const float* __restrict__ p2,
                    float* __restrict__ out_pred,
                    float* __restrict__ out_dist) {
    __shared__ float s_p2[CDIM];
    __shared__ __attribute__((aligned(16))) unsigned short s_phi[2][TILE_C * ROW_US];

    for (int i = threadIdx.x; i < CDIM; i += blockDim.x) s_p2[i] = p2[i];

    const int wave = threadIdx.x >> 5;
    const int lane = threadIdx.x & 31;
    const int m = lane & 15;                 // tile row / centroid lane
    const int h = lane >> 4;                 // lane half

    const long rowbase = ((long)blockIdx.x * WAVES_PER_BLOCK + wave) * 16;
    const bool active = rowbase < NPTS;      // idle tail waves still barrier
    const long rb = active ? rowbase : 0;

    // Kick off the first TDM staging transfer.
    if (wave == 0)
        tdm_load_tile(phi_bf, (unsigned)(uintptr_t)&s_phi[0][0]);

    // ---- A fragments (contiguous k = 16h + s per lane half, same order as B), ||x||^2 fused into the
    // f32->bf16 conversion so X is read exactly once.
    union AFrag { v16bf v; unsigned short u[16]; };
    AFrag a[8];
    const float* xrow = X + (size_t)(rb + m) * DDIM;
    float ssq = 0.0f;
#pragma unroll
    for (int kk = 0; kk < 8; ++kk) {
        const int kb = kk * 32;
        const float4* g0 = (const float4*)(xrow + kb + 16 * h);
        const float4* g1 = (const float4*)(xrow + kb + 16 * h + 8);
        const float4 f0 = g0[0], f1 = g0[1], f2 = g1[0], f3 = g1[1];
        float t[16] = { f0.x, f0.y, f0.z, f0.w,  f1.x, f1.y, f1.z, f1.w,
                        f2.x, f2.y, f2.z, f2.w,  f3.x, f3.y, f3.z, f3.w };
#pragma unroll
        for (int i = 0; i < 16; ++i) {
            ssq = fmaf(t[i], t[i], ssq);
            a[kk].u[i] = f32_to_bf16_rne(t[i]);
        }
    }

    float best[8];
    int   bcol[8];
#pragma unroll
    for (int r = 0; r < 8; ++r) { best[r] = 3.402823e38f; bcol[r] = 0; }

    union BFrag { v16bf v; uint4 q[2]; };

    for (int ct = 0; ct < NTILES; ++ct) {
        if (wave == 0) {
            // Prefetch next tile into the other buffer (clamped re-load of the
            // last tile keeps the wait count constant; target buffer is idle).
            const int nxt = (ct + 1 < NTILES) ? ct + 1 : NTILES - 1;
            tdm_load_tile(phi_bf + (size_t)nxt * TILE_C * DDIM,
                          (unsigned)(uintptr_t)&s_phi[(ct + 1) & 1][0]);
            __builtin_amdgcn_s_wait_tensorcnt(1);   // tile ct is in LDS
        }
        __syncthreads();                            // release consumers

        const unsigned short* sb = &s_phi[ct & 1][0];
#pragma unroll
        for (int sub = 0; sub < 2; ++sub) {
            const int cb = ct * TILE_C + sub * 16;
            // B = Phi^T fragment from padded LDS rows: lane m -> centroid
            // cb+m, K contiguous; 528 B pitch => ds_load_b128 hits all 64 banks.
            const unsigned short* prow = sb + (sub * 16 + m) * ROW_US + 16 * h;

            v8f acc0 = {0.f,0.f,0.f,0.f,0.f,0.f,0.f,0.f};
            v8f acc1 = {0.f,0.f,0.f,0.f,0.f,0.f,0.f,0.f};
#pragma unroll
            for (int kk = 0; kk < 8; kk += 2) {      // two independent chains
                BFrag b0, b1;
                const uint4* bp0 = (const uint4*)(prow + kk * 32);
                const uint4* bp1 = (const uint4*)(prow + (kk + 1) * 32);
                b0.q[0] = bp0[0]; b0.q[1] = bp0[1];
                b1.q[0] = bp1[0]; b1.q[1] = bp1[1];
                acc0 = __builtin_amdgcn_wmma_f32_16x16x32_bf16(
                           false, a[kk].v,     false, b0.v, (short)0, acc0, false, false);
                acc1 = __builtin_amdgcn_wmma_f32_16x16x32_bf16(
                           false, a[kk + 1].v, false, b1.v, (short)0, acc1, false, false);
            }

            const float p2v = s_p2[cb + m];
#pragma unroll
            for (int r = 0; r < 8; ++r) {
                const float sc = fmaf(-2.0f, acc0[r] + acc1[r], p2v);
                if (sc < best[r]) { best[r] = sc; bcol[r] = cb + m; }
            }
        }
        __syncthreads();    // all waves done reading before buffer is reused
    }

    // ||x||^2: lane halves (m, m+16) together hold row m's 256 values.
    const float x2full = ssq + __shfl_xor(ssq, 16, 32);

    // argmin across the 16 lanes of each half (tie-break: lowest index).
#pragma unroll
    for (int r = 0; r < 8; ++r) {
#pragma unroll
        for (int off = 8; off >= 1; off >>= 1) {
            const float ob = __shfl_xor(best[r], off, 32);
            const int   oc = __shfl_xor(bcol[r], off, 32);
            if (ob < best[r] || (ob == best[r] && oc < bcol[r])) {
                best[r] = ob; bcol[r] = oc;
            }
        }
    }

    // Writer lanes (m==0 per half) need x2 of rows 8h..8h+7 (lanes 8h..8h+7);
    // gather with all lanes active (EXEC rules for ds_bpermute).
    float x2r[8];
#pragma unroll
    for (int r = 0; r < 8; ++r) x2r[r] = __shfl(x2full, (h << 3) + r, 32);

    if (active && m == 0) {
#pragma unroll
        for (int r = 0; r < 8; ++r) {
            const long row = rowbase + (h << 3) + r;
            out_pred[row] = (float)bcol[r];
            out_dist[row] = sqrtf(fmaxf(x2r[r] + best[r], 0.0f));
        }
    }
}

// ---------------------------------------------------------------------------
extern "C" void kernel_launch(void* const* d_in, const int* in_sizes, int n_in,
                              void* d_out, int out_size, void* d_ws, size_t ws_size,
                              hipStream_t stream) {
    (void)in_sizes; (void)n_in; (void)out_size; (void)ws_size;

    const float* X   = (const float*)d_in[0];   // [N, D] f32
    const float* mp  = (const float*)d_in[1];   // [C, C] f32
    const float* eps = (const float*)d_in[2];   // [C, D] f32

    // workspace: Phi bf16 [C, D] (512 KB) + p2 f32 [C] (4 KB)
    unsigned short* phi_bf = (unsigned short*)d_ws;
    float* p2 = (float*)((char*)d_ws + (size_t)CDIM * DDIM * sizeof(unsigned short));

    float* out_pred = (float*)d_out;            // pred as float(index)
    float* out_dist = out_pred + NPTS;          // min distances

    phi_p2_kernel<<<CDIM / PHI_ROWS, 256, 0, stream>>>(mp, eps, phi_bf, p2);

    const int total_waves = NPTS / 16;          // 6250
    const int nblocks = (total_waves + WAVES_PER_BLOCK - 1) / WAVES_PER_BLOCK;
    cdist_argmin_kernel<<<nblocks, 32 * WAVES_PER_BLOCK, 0, stream>>>(
        X, phi_bf, p2, out_pred, out_dist);
}
